// CWTLayer_65506841198870
// MI455X (gfx1250) — matrix operand order repaired and checked by reading, not compiled, not run on hardware
//
#include <hip/hip_runtime.h>

typedef _Float16 v16h __attribute__((ext_vector_type(16)));
typedef _Float16 v8h  __attribute__((ext_vector_type(8)));
typedef float    v8f  __attribute__((ext_vector_type(8)));

#define NB    32
#define NT    4096
#define NC    3
#define NSIG  96
#define NSC   128
#define NPSI  4096
#define XPS   8192          // padded-signal row stride (halfs), actual data 8190
#define GTS   3328          // Gt row stride (halfs) == max Kpad
#define STEPF (16.0f / 4095.0f)

// CDNA5 async global->LDS path (ASYNCcnt), with synchronous fallback
#if __has_builtin(__builtin_amdgcn_global_load_async_to_lds_b128)
#define USE_ASYNC_LDS 1
// builtin prototype is (v4i addrspace(1)*, v4i addrspace(3)*, imm int, imm int)
typedef int v4i_ __attribute__((vector_size(4 * sizeof(int))));
typedef __attribute__((address_space(1))) v4i_ g_v4i;   // global
typedef __attribute__((address_space(3))) v4i_ l_v4i;   // LDS
#else
#define USE_ASYNC_LDS 0
#endif

__device__ __forceinline__ void wait_async_all() {
#if USE_ASYNC_LDS
#if __has_builtin(__builtin_amdgcn_s_wait_asynccnt)
  __builtin_amdgcn_s_wait_asynccnt(0);
#else
  asm volatile("s_wait_asynccnt 0" ::: "memory");
#endif
#endif
}

// scales: logspace(log10(2), log10(204), 128)  (geometric 2 .. 204)
__device__ __forceinline__ float scale_of(int i) {
  // log2(204) = 7.67242534
  return exp2f(1.0f + (float)i * ((7.67242534f - 1.0f) / 127.0f));
}

// ---------------------------------------------------------------------------
// 1) reflect-pad x (B,T,C) -> f16 xph[sig=c*32+b][8192]
// ---------------------------------------------------------------------------
__global__ __launch_bounds__(256) void k_pad(const float* __restrict__ x,
                                             _Float16* __restrict__ xph) {
  const int sig = blockIdx.y;
  const int tp  = blockIdx.x * 256 + threadIdx.x;       // 0..8191
  const int c = sig / NB, b = sig % NB;
  float v = 0.0f;
  if (tp < NT + 2 * 2047) {                              // 8190 valid samples
    int t = tp - 2047;
    if (t < 0) t = -t;
    if (t > NT - 1) t = 2 * (NT - 1) - t;
    v = x[((size_t)b * NT + t) * NC + c];
  }
  xph[(size_t)sig * XPS + tp] = (_Float16)v;
}

// ---------------------------------------------------------------------------
// 2) build diffed, flipped, sqrt(s)-scaled, center-aligned kernels Gt[128][GTS]
//    coef[t] = sum_u xp[t + dgu + u] * Gt[s][u]
// ---------------------------------------------------------------------------
__device__ __forceinline__ float kern_at(const float* psi, int k, int K, float inv) {
  if (k < 0 || k >= K) return 0.0f;
  const int i = K - 1 - k;                               // flip
  int j = (int)floorf((float)i * inv);
  j = min(max(j, 0), NPSI - 1);
  return psi[j];
}

__global__ __launch_bounds__(256) void k_gkern(const float* __restrict__ psi,
                                               _Float16* __restrict__ Gt) {
  const int u  = blockIdx.x * 256 + threadIdx.x;         // 0..GTS-1
  const int si = blockIdx.y;                             // scale index
  if (u >= GTS) return;
  const float s = scale_of(si);
  const int K = (int)ceilf(s * 16.0f + 1.0f);
  const int d = (4094 - K) >> 1;
  // group geometry (aligned, 8-rounded base offset)
  const float sgv = scale_of((si >> 4) * 16 + 15);
  const int Kg  = (int)ceilf(sgv * 16.0f + 1.0f);
  const int dgu = ((4094 - Kg) >> 1) & ~7;
  const int up  = u - (d - dgu);                         // position in [0, K]
  float val = 0.0f;
  if (up >= 0 && up <= K) {
    const float inv = 1.0f / (s * STEPF);
    val = -sqrtf(s) * (kern_at(psi, up - 1, K, inv) - kern_at(psi, up, K, inv));
  }
  Gt[(size_t)si * GTS + u] = (_Float16)val;
}

// ---------------------------------------------------------------------------
// 3) main WMMA GEMM: per (sig, 256-time tile, scale-group of 16)
//    coef_h[sig][scale][t] f16, M=256 x N=16 x K=Kpad(group)
// ---------------------------------------------------------------------------
__global__ __launch_bounds__(256) void k_cwt(const _Float16* __restrict__ xph,
                                             const _Float16* __restrict__ Gt,
                                             _Float16* __restrict__ coef) {
  __shared__ __align__(16) _Float16 lA[2][384];          // Hankel window tile
  __shared__ __align__(16) _Float16 lB[2][16 * 128];     // 16 scales x 128 K

  const int tid = threadIdx.x;
  const int sig = blockIdx.y;
  const int g   = blockIdx.z;
  const int t0  = blockIdx.x * 256;

  // group geometry (recomputed cheaply; matches k_gkern)
  const float sg = scale_of(g * 16 + 15);
  const int Kg   = (int)ceilf(sg * 16.0f + 1.0f);
  const int dg   = (4094 - Kg) >> 1;
  const int dgu  = dg & ~7;
  const int Kpad = (Kg + 1 + (dg - dgu) + 31) & ~31;

  const _Float16* xrow = xph + (size_t)sig * XPS + t0 + dgu;   // 16B aligned
  const _Float16* grow = Gt + (size_t)(g * 16) * GTS;

  const int lane = tid & 31;
  const int wv   = tid >> 5;        // wave 0..7  -> M rows [wv*32, wv*32+32)
  const int mrow = lane & 15;
  const int hi   = lane >> 4;

  v8f acc0 = {};
  v8f acc1 = {};

  auto stage = [&](int kofs, int buf) {
#if USE_ASYNC_LDS
    // async copy engine: no VGPR round-trip, tracked on ASYNCcnt
    if (tid < 48) {                                      // 48 * 16B = 384 halfs
      __builtin_amdgcn_global_load_async_to_lds_b128(
          (g_v4i*)(xrow + kofs + tid * 8),
          (l_v4i*)&lA[buf][tid * 8], 0, 0);
    }
    {
      const int row = tid >> 4, c8 = tid & 15;           // 256 * 16B = 16x128 halfs
      __builtin_amdgcn_global_load_async_to_lds_b128(
          (g_v4i*)(grow + (size_t)row * GTS + kofs + c8 * 8),
          (l_v4i*)&lB[buf][row * 128 + c8 * 8], 0, 0);
    }
#else
    if (tid < 48) {
      ((int4*)lA[buf])[tid] = ((const int4*)(xrow + kofs))[tid];
      __builtin_prefetch((const void*)(xrow + kofs + 128 + tid * 8), 0, 1);
    }
    const int row = tid >> 4, c8 = tid & 15;
    ((int4*)(&lB[buf][row * 128]))[c8] =
        ((const int4*)(grow + (size_t)row * GTS + kofs))[c8];
#endif
  };

  auto compute = [&](int buf) {
#pragma unroll
    for (int ks = 0; ks < 4; ++ks) {                     // 4 x K=32 substeps
      // B fragment: lane<16 -> col mrow, K=ks*32+0..7,16..23 ; lane>=16 -> K=+8..15,24..31
      v16h bf;
      {
        const v8h b0 = *(const v8h*)&lB[buf][mrow * 128 + ks * 32 + hi * 8];
        const v8h b1 = *(const v8h*)&lB[buf][mrow * 128 + ks * 32 + hi * 8 + 16];
#pragma unroll
        for (int i = 0; i < 8; ++i) { bf[i] = b0[i]; bf[i + 8] = b1[i]; }
      }
#pragma unroll
      for (int p = 0; p < 2; ++p) {                      // two 16-row M tiles
        // A fragment (Hankel): row m = wv*32+p*16+mrow, A[m][k]=lA[m+k]
        // lanes<16: halves 0..7 = K0..7, 8..15 = K16..23 (+8 for upper lanes)
        const int ab = wv * 32 + p * 16 + mrow + ks * 32 + hi * 8;
        const v8h a0 = *(const v8h*)&lA[buf][ab];
        const v8h a1 = *(const v8h*)&lA[buf][ab + 16];
        v16h af;
#pragma unroll
        for (int i = 0; i < 8; ++i) { af[i] = a0[i]; af[i + 8] = a1[i]; }
        if (p == 0)
          acc0 = __builtin_amdgcn_wmma_f32_16x16x32_f16(
              false, af, false, bf, (short)0, acc0, false, false);
        else
          acc1 = __builtin_amdgcn_wmma_f32_16x16x32_f16(
              false, af, false, bf, (short)0, acc1, false, false);
      }
    }
  };

  int buf = 0;
  stage(0, 0);
  wait_async_all();
  __syncthreads();
  for (int kofs = 0;;) {
    const int nxt = kofs + 128;
    if (nxt < Kpad) stage(nxt, buf ^ 1);                 // overlap copy w/ WMMA
    compute(buf);
    kofs = nxt;
    if (kofs >= Kpad) break;
    buf ^= 1;
    wait_async_all();                                    // drain ASYNCcnt
    __syncthreads();                                     // publish staged buffer
  }

  // D layout: VGPR r -> M = r (lanes<16) / r+8 (lanes>=16); N = mrow
  const int tbase = t0 + wv * 32 + hi * 8;
  _Float16* orow = coef + ((size_t)sig * NSC + g * 16 + mrow) * NT;
  v8h h0, h1;
#pragma unroll
  for (int r = 0; r < 8; ++r) { h0[r] = (_Float16)acc0[r]; h1[r] = (_Float16)acc1[r]; }
  *(v8h*)(orow + tbase)      = h0;   // tile p=0, 8 consecutive t
  *(v8h*)(orow + tbase + 16) = h1;   // tile p=1
}

// ---------------------------------------------------------------------------
// 4) bilinear resize (B,128,4096,C) -> (B,224,224,C); antialiased along t
// ---------------------------------------------------------------------------
__global__ __launch_bounds__(256) void k_resize(const _Float16* __restrict__ coef,
                                                float* __restrict__ out) {
  const int idx = blockIdx.x * 256 + threadIdx.x;
  const int total = NB * 224 * 224 * NC;
  if (idx >= total) return;
  const int c = idx % NC;
  const int w = (idx / NC) % 224;
  const int h = (idx / (NC * 224)) % 224;
  const int b = idx / (NC * 224 * 224);

  // scales dim: 128 -> 224 (upsample, plain linear)
  const float fy = (h + 0.5f) * (128.0f / 224.0f) - 0.5f;
  int y0 = (int)floorf(fy);
  const float wy = fy - (float)y0;
  const int y1 = min(y0 + 1, NSC - 1);
  y0 = max(y0, 0);

  // time dim: 4096 -> 224 (downsample, triangle kernel widened by ratio)
  const float r  = 4096.0f / 224.0f;
  const float cx = (w + 0.5f) * r - 0.5f;
  const int x0 = (int)ceilf(cx - r);
  const int x1 = (int)floorf(cx + r);

  const _Float16* row0 = coef + ((size_t)(c * NB + b) * NSC + y0) * NT;
  const _Float16* row1 = coef + ((size_t)(c * NB + b) * NSC + y1) * NT;

  float a0 = 0.0f, a1 = 0.0f, wsum = 0.0f;
  for (int t = x0; t <= x1; ++t) {
    const float wt = 1.0f - fabsf((float)t - cx) * (1.0f / r);
    if (wt <= 0.0f) continue;
    const int tc = min(max(t, 0), NT - 1);
    wsum += wt;
    a0 += wt * (float)row0[tc];
    a1 += wt * (float)row1[tc];
  }
  out[idx] = ((1.0f - wy) * a0 + wy * a1) / wsum;
}

// ---------------------------------------------------------------------------
extern "C" void kernel_launch(void* const* d_in, const int* in_sizes, int n_in,
                              void* d_out, int out_size, void* d_ws, size_t ws_size,
                              hipStream_t stream) {
  const float* x   = (const float*)d_in[0];   // (32,4096,3)
  const float* psi = (const float*)d_in[1];   // (4096,)
  float* out = (float*)d_out;                 // (32,224,224,3)

  char* ws = (char*)d_ws;
  const size_t off_xph = 0;
  const size_t off_gt  = off_xph + (size_t)NSIG * XPS * sizeof(_Float16); // 1.5 MB
  const size_t off_cf  = off_gt  + (size_t)NSC * GTS * sizeof(_Float16);  // +0.85 MB
  _Float16* xph  = (_Float16*)(ws + off_xph);
  _Float16* gt   = (_Float16*)(ws + off_gt);
  _Float16* coef = (_Float16*)(ws + off_cf);  // 96*128*4096 f16 = 100.7 MB

  k_pad   <<<dim3(XPS / 256, NSIG), 256, 0, stream>>>(x, xph);
  k_gkern <<<dim3(GTS / 256, NSC), 256, 0, stream>>>(psi, gt);
  k_cwt   <<<dim3(NT / 256, NSIG, 8), 256, 0, stream>>>(xph, gt, coef);
  const int total = NB * 224 * 224 * NC;
  k_resize<<<(total + 255) / 256, 256, 0, stream>>>(coef, out);
}
